// MultiHeadAttention_80126909874523
// MI455X (gfx1250) — matrix operand order repaired and checked by reading, not compiled, not run on hardware
//
#include <hip/hip_runtime.h>
#include <hip/hip_bf16.h>

// ---------------------------------------------------------------------------
// MI455X (gfx1250) multi-head attention, bf16 WMMA + async/TDM data movement.
//   B=4, T=2048, C=512, H=8, D=64, INNER=512, OUT=512
// ---------------------------------------------------------------------------

typedef __bf16 bf16_t;
typedef __bf16 bf16x8  __attribute__((ext_vector_type(8)));
typedef __bf16 bf16x16 __attribute__((ext_vector_type(16)));
typedef float  f32x8   __attribute__((ext_vector_type(8)));
typedef int      i32x4 __attribute__((ext_vector_type(4)));
typedef int      i32x8 __attribute__((ext_vector_type(8)));
typedef unsigned u32x4 __attribute__((ext_vector_type(4)));
typedef int      v4i_vs __attribute__((vector_size(16)));  // matches builtin

#define NUM_B 4
#define SEQ_T 2048
#define CDIM  512
#define NHEAD 8
#define HDIM  64
#define IDIM  512

#if defined(__has_builtin)
#if __has_builtin(__builtin_amdgcn_global_load_async_to_lds_b128)
#define HAVE_ASYNC_LDS 1
#else
#define HAVE_ASYNC_LDS 0
#endif
#if __has_builtin(__builtin_amdgcn_tensor_load_to_lds)
#define HAVE_TDM 1
#else
#define HAVE_TDM 0
#endif
#else
#define HAVE_ASYNC_LDS 0
#define HAVE_TDM 0
#endif

// Low 32 bits of a generic shared pointer == wave-relative LDS byte offset.
__device__ __forceinline__ unsigned lds_offset(const void* p) {
  return (unsigned)(uintptr_t)p;
}

// 16-byte global -> LDS copy; async (ASYNCcnt) when available.
__device__ __forceinline__ void async_cp16(const bf16_t* g, bf16_t* l) {
#if HAVE_ASYNC_LDS
  __builtin_amdgcn_global_load_async_to_lds_b128(
      (__attribute__((address_space(1))) v4i_vs*)(uintptr_t)g,
      (__attribute__((address_space(3))) v4i_vs*)lds_offset(l), 0, 0);
#else
  *(bf16x8*)l = *(const bf16x8*)g;
#endif
}

__device__ __forceinline__ void wait_async0() {
#if HAVE_ASYNC_LDS
#if __has_builtin(__builtin_amdgcn_s_wait_asynccnt)
  __builtin_amdgcn_s_wait_asynccnt(0);
#else
  asm volatile("s_wait_asynccnt 0" ::: "memory");
#endif
#endif
}

#if HAVE_TDM
// TDM: DMA one contiguous 4KB block (32 keys x 64 d of bf16) into LDS as a
// 1-D tile of 512 x 8-byte units. D# layout per CDNA5 ISA sect. 8.3/8.4.
__device__ __forceinline__ void tdm_load_4kb(bf16_t* lds_dst,
                                             const bf16_t* gsrc) {
  unsigned long long ga = (unsigned long long)(uintptr_t)gsrc;
  u32x4 g0;
  g0[0] = 1u;                                            // count=1, user mode
  g0[1] = lds_offset(lds_dst);                           // lds_addr (bytes)
  g0[2] = (unsigned)(ga & 0xFFFFFFFFu);                  // global_addr[31:0]
  g0[3] = (unsigned)((ga >> 32) & 0x1FFFFFFu) | (2u << 30);  // [56:32]|type=2
  i32x8 g1;
  g1[0] = 0x00030000;         // data_size=3 (8B units); no mask/pad/iterate
  g1[1] = (int)(512u << 16);  // tensor_dim0 = 512 units (bits [63:48])
  g1[2] = (int)(1u << 16);    // tensor_dim1 = 1       (bits [95:80])
  g1[3] = (int)(512u << 16);  // tile_dim0  = 512      (bits [127:112])
  g1[4] = 0;                  // tile_dim1 = tile_dim2 = 0 (unused)
  g1[5] = 512;                // tensor_dim0_stride = 512 units
  g1[6] = 0;
  g1[7] = 0;
  i32x4 z4 = {};
#if __clang_major__ >= 23
  i32x8 z8 = {};
  __builtin_amdgcn_tensor_load_to_lds(g0, g1, z4, z4, z8, 0);
#else
  __builtin_amdgcn_tensor_load_to_lds(g0, g1, z4, z4, 0);
#endif
}
#endif

// Build a 16-element bf16 A/B fragment from two contiguous 8-element halves.
__device__ __forceinline__ bf16x16 frag2(const bf16_t* p0, const bf16_t* p1) {
  bf16x8 lo = *(const bf16x8*)p0;
  bf16x8 hi = *(const bf16x8*)p1;
  bf16x16 r;
#pragma unroll
  for (int i = 0; i < 8; ++i) { r[i] = lo[i]; r[i + 8] = hi[i]; }
  return r;
}

__device__ __forceinline__ f32x8 wmma_bf16(bf16x16 a, bf16x16 b, f32x8 c) {
  // D = A(16x32 bf16) * B(32x16 bf16) + C(16x16 f32)
  return __builtin_amdgcn_wmma_f32_16x16x32_bf16(false, a, false, b,
                                                 (short)0, c, false, false);
}

// ---------------------------------------------------------------------------
// Kernel 0: fp32 -> bf16 conversion (grid-stride).
// ---------------------------------------------------------------------------
__global__ void mha_cvt_bf16_kernel(const float* __restrict__ src,
                                    bf16_t* __restrict__ dst, int n) {
  for (int i = blockIdx.x * blockDim.x + threadIdx.x; i < n;
       i += gridDim.x * blockDim.x)
    dst[i] = (bf16_t)src[i];
}

// ---------------------------------------------------------------------------
// Kernel 1: fused QKV projection GEMM.
//   M = B*T = 8192, N = 3*512 (Wq|Wk|Wv stacked), K = 512.
//   Q,K written as (B,H,T,D); V written transposed as (B,H,D,T).
// ---------------------------------------------------------------------------
__global__ __launch_bounds__(128) void mha_qkv_gemm_kernel(
    const bf16_t* __restrict__ Xb, const bf16_t* __restrict__ Wstk,
    bf16_t* __restrict__ Qb, bf16_t* __restrict__ Kb, bf16_t* __restrict__ Vb) {
  const int wave = threadIdx.x >> 5, lane = threadIdx.x & 31;
  const int lh = lane >> 4, ln = lane & 15;
  const int mrow = blockIdx.x * 64 + wave * 16;
  const int ncol0 = blockIdx.y * 128;

  f32x8 acc[8] = {};
  const bf16_t* arow = Xb + (size_t)(mrow + ln) * CDIM;
#pragma unroll 4
  for (int kc = 0; kc < CDIM / 32; ++kc) {
    const bf16_t* ap = arow + kc * 32;
    bf16x16 a = frag2(ap + lh * 16, ap + lh * 16 + 8);
#pragma unroll
    for (int nt = 0; nt < 8; ++nt) {
      const bf16_t* bp =
          Wstk + (size_t)(ncol0 + nt * 16 + ln) * CDIM + kc * 32 + lh * 16;
      acc[nt] = wmma_bf16(a, frag2(bp, bp + 8), acc[nt]);
    }
  }

  const float cs = 0.044194173824159216f;  // 512^-0.5
#pragma unroll
  for (int nt = 0; nt < 8; ++nt) {
    const int col = ncol0 + nt * 16 + ln;
#pragma unroll
    for (int r = 0; r < 8; ++r) {
      const int row = mrow + r + 8 * lh;
      const int b = row >> 11, t = row & (SEQ_T - 1);
      const float v = acc[nt][r] * cs;
      if (col < 512) {
        const int hh = col >> 6, d = col & 63;
        Qb[((size_t)(b * NHEAD + hh) * SEQ_T + t) * HDIM + d] = (bf16_t)v;
      } else if (col < 1024) {
        const int c2 = col - 512, hh = c2 >> 6, d = c2 & 63;
        Kb[((size_t)(b * NHEAD + hh) * SEQ_T + t) * HDIM + d] = (bf16_t)v;
      } else {
        const int c2 = col - 1024, hh = c2 >> 6, d = c2 & 63;
        Vb[((size_t)(b * NHEAD + hh) * HDIM + d) * SEQ_T + t] = (bf16_t)v;
      }
    }
  }
}

// ---------------------------------------------------------------------------
// Kernel 2: flash attention (online softmax), 4 waves x 16 queries each.
//   K block staged by the Tensor Data Mover (TENSORcnt); V^T block staged by
//   per-lane GLOBAL_LOAD_ASYNC_TO_LDS_B128 (ASYNCcnt). P tile round-trips
//   per-wave LDS for C-layout -> A-layout.
// ---------------------------------------------------------------------------
__global__ __launch_bounds__(128) void mha_flash_attn_kernel(
    const bf16_t* __restrict__ Qb, const bf16_t* __restrict__ Kb,
    const bf16_t* __restrict__ Vb, bf16_t* __restrict__ Ab) {
  __shared__ __align__(16) bf16_t sK[32 * 64];    // [key][d]
  __shared__ __align__(16) bf16_t sV[64 * 32];    // [d][key]
  __shared__ __align__(16) bf16_t sP[4][16 * 32]; // per-wave P tile

  const int tid = threadIdx.x;
  const int wave = tid >> 5, lane = tid & 31;
  const int lh = lane >> 4, ln = lane & 15;
  const int bh = blockIdx.y;
  const int b = bh >> 3, head = bh & 7;
  const int q0 = blockIdx.x * 64 + wave * 16;

  const bf16_t* Qp = Qb + (size_t)bh * SEQ_T * HDIM;
  const bf16_t* Kp = Kb + (size_t)bh * SEQ_T * HDIM;
  const bf16_t* Vp = Vb + (size_t)bh * HDIM * SEQ_T;

  // Q fragments held in registers for the whole key loop (16 x 64 = 2 chunks).
  bf16x16 qf[2];
#pragma unroll
  for (int ch = 0; ch < 2; ++ch) {
    const bf16_t* qp = Qp + (size_t)(q0 + ln) * HDIM + ch * 32;
    qf[ch] = frag2(qp + lh * 16, qp + lh * 16 + 8);
  }

  float mrow[8], lrow[8];
#pragma unroll
  for (int r = 0; r < 8; ++r) { mrow[r] = -1e30f; lrow[r] = 0.f; }
  f32x8 Oacc[4] = {};

  const float dscale = 0.125f;  // 64^-0.5
  const int nblk = SEQ_T / 32;
  for (int kb = 0; kb < nblk; ++kb) {
    __syncthreads();
#if HAVE_TDM
    if (wave == 0) {  // K block: one TDM DMA of the whole 4 KB tile
      tdm_load_4kb(sK, Kp + (size_t)kb * 32 * HDIM);
    }
#else
    {  // K block: 32 keys x 64 d (4 KB), 16 elements per thread
      const bf16_t* g = Kp + (size_t)kb * 32 * HDIM + tid * 16;
      bf16_t* s = sK + tid * 16;
      async_cp16(g, s);
      async_cp16(g + 8, s + 8);
    }
#endif
    {  // V^T block: 64 d x 32 keys, async to LDS
      const int vd = tid >> 1, vo = (tid & 1) * 16;
      const bf16_t* g = Vp + (size_t)vd * SEQ_T + kb * 32 + vo;
      bf16_t* s = sV + vd * 32 + vo;
      async_cp16(g, s);
      async_cp16(g + 8, s + 8);
    }
    if (kb + 1 < nblk) {  // global_prefetch_b8 for the next block
      __builtin_prefetch(Kp + (size_t)(kb + 1) * 32 * HDIM + tid * 16, 0, 3);
      __builtin_prefetch(Vp + (size_t)(tid >> 1) * SEQ_T + (kb + 1) * 32 +
                             (tid & 1) * 16, 0, 3);
    }
    wait_async0();
#if HAVE_TDM
    if (wave == 0) __builtin_amdgcn_s_wait_tensorcnt(0);
#endif
    __syncthreads();

    // S = Q K^T  (16 q x 32 keys = 2 N-tiles, K-dim = 64 = 2 chunks)
    f32x8 sc[2] = {};
#pragma unroll
    for (int ch = 0; ch < 2; ++ch) {
#pragma unroll
      for (int nt = 0; nt < 2; ++nt) {
        const bf16_t* bp = sK + (nt * 16 + ln) * 64 + ch * 32 + lh * 16;
        sc[nt] = wmma_bf16(qf[ch], frag2(bp, bp + 8), sc[nt]);
      }
    }

    // Online softmax. C-layout: lane holds a column; rows r+8*lh -> row
    // reductions are shfl_xor over the 16-lane half-group (masks 1,2,4,8).
    float alpha[8];
#pragma unroll
    for (int r = 0; r < 8; ++r) {
      float s0 = sc[0][r] * dscale, s1 = sc[1][r] * dscale;
      float mx = fmaxf(s0, s1);
#pragma unroll
      for (int msk = 1; msk < 16; msk <<= 1)
        mx = fmaxf(mx, __shfl_xor(mx, msk, 32));
      const float mnew = fmaxf(mrow[r], mx);
      alpha[r] = __expf(mrow[r] - mnew);
      const float p0 = __expf(s0 - mnew);
      const float p1 = __expf(s1 - mnew);
      sc[0][r] = p0; sc[1][r] = p1;
      float sum = p0 + p1;
#pragma unroll
      for (int msk = 1; msk < 16; msk <<= 1)
        sum += __shfl_xor(sum, msk, 32);
      lrow[r] = lrow[r] * alpha[r] + sum;
      mrow[r] = mnew;
    }
#pragma unroll
    for (int dt = 0; dt < 4; ++dt)
#pragma unroll
      for (int r = 0; r < 8; ++r) Oacc[dt][r] *= alpha[r];

    // P (C-layout) -> per-wave LDS [q 16][key 32] -> A-fragment.
    bf16_t* pw = sP[wave];
#pragma unroll
    for (int r = 0; r < 8; ++r) {
      const int row = r + 8 * lh;
      pw[row * 32 + ln]      = (bf16_t)sc[0][r];
      pw[row * 32 + 16 + ln] = (bf16_t)sc[1][r];
    }
    asm volatile("s_wait_dscnt 0" ::: "memory");  // wave-local store->load
    bf16x16 pf = frag2(pw + ln * 32 + lh * 16, pw + ln * 32 + lh * 16 + 8);

    // O += P @ V  (K-dim = 32 keys, 4 d-tiles)
#pragma unroll
    for (int dt = 0; dt < 4; ++dt) {
      const bf16_t* vp = sV + (dt * 16 + ln) * 32 + lh * 16;
      Oacc[dt] = wmma_bf16(pf, frag2(vp, vp + 8), Oacc[dt]);
    }
  }

  // Normalize and write attention output (B,T,INNER) bf16.
  float inv[8];
#pragma unroll
  for (int r = 0; r < 8; ++r) inv[r] = 1.0f / lrow[r];
#pragma unroll
  for (int dt = 0; dt < 4; ++dt) {
#pragma unroll
    for (int r = 0; r < 8; ++r) {
      const int t = q0 + r + 8 * lh;
      const int i = head * HDIM + dt * 16 + ln;
      Ab[((size_t)(b * SEQ_T + t)) * IDIM + i] = (bf16_t)(Oacc[dt][r] * inv[r]);
    }
  }
}

// ---------------------------------------------------------------------------
// Kernel 3: output projection GEMM. M=8192, N=512, K=512, fp32 stores.
// ---------------------------------------------------------------------------
__global__ __launch_bounds__(128) void mha_out_proj_kernel(
    const bf16_t* __restrict__ Ab, const bf16_t* __restrict__ Wpb,
    float* __restrict__ out) {
  const int wave = threadIdx.x >> 5, lane = threadIdx.x & 31;
  const int lh = lane >> 4, ln = lane & 15;
  const int mrow = blockIdx.x * 64 + wave * 16;
  const int ncol0 = blockIdx.y * 128;

  f32x8 acc[8] = {};
  const bf16_t* arow = Ab + (size_t)(mrow + ln) * IDIM;
#pragma unroll 4
  for (int kc = 0; kc < IDIM / 32; ++kc) {
    const bf16_t* ap = arow + kc * 32;
    bf16x16 a = frag2(ap + lh * 16, ap + lh * 16 + 8);
#pragma unroll
    for (int nt = 0; nt < 8; ++nt) {
      const bf16_t* bp =
          Wpb + (size_t)(ncol0 + nt * 16 + ln) * IDIM + kc * 32 + lh * 16;
      acc[nt] = wmma_bf16(a, frag2(bp, bp + 8), acc[nt]);
    }
  }

  const float ps = 0.044194173824159216f;  // 512^-0.5
#pragma unroll
  for (int nt = 0; nt < 8; ++nt) {
    const int col = ncol0 + nt * 16 + ln;
#pragma unroll
    for (int r = 0; r < 8; ++r) {
      const int row = mrow + r + 8 * lh;
      out[(size_t)row * 512 + col] = acc[nt][r] * ps;
    }
  }
}

// ---------------------------------------------------------------------------
// Launch. Inputs: x, Wk, Wq, Wv, Wp, causal_mask(=0, ignored).
// Workspace (bf16 elements):
//   Qb[4M] | Kb[4M] | Vb[4M] | Wq|Wk|Wv[768K] | Wp[256K] | Ab[4M] | Xb[4M]
//   total ~44 MB (fits easily in L2's 192 MB -> compute-bound by design).
// ---------------------------------------------------------------------------
extern "C" void kernel_launch(void* const* d_in, const int* in_sizes, int n_in,
                              void* d_out, int out_size, void* d_ws,
                              size_t ws_size, hipStream_t stream) {
  (void)in_sizes; (void)n_in; (void)out_size; (void)ws_size;
  const float* x  = (const float*)d_in[0];
  const float* Wk = (const float*)d_in[1];
  const float* Wq = (const float*)d_in[2];
  const float* Wv = (const float*)d_in[3];
  const float* Wp = (const float*)d_in[4];
  float* out = (float*)d_out;

  bf16_t* ws = (bf16_t*)d_ws;
  const size_t QKV = (size_t)NUM_B * NHEAD * SEQ_T * HDIM;  // 4,194,304
  bf16_t* Qb   = ws;
  bf16_t* Kb   = Qb + QKV;
  bf16_t* Vb   = Kb + QKV;
  bf16_t* Wstk = Vb + QKV;                 // [Wq;Wk;Wv], 3*512*512
  bf16_t* Wpb  = Wstk + 3 * 512 * 512;
  bf16_t* Ab   = Wpb + 512 * 512;          // (B*T, 512) bf16
  bf16_t* Xb   = Ab + (size_t)NUM_B * SEQ_T * IDIM;

  mha_cvt_bf16_kernel<<<2048, 256, 0, stream>>>(x, Xb, NUM_B * SEQ_T * CDIM);
  mha_cvt_bf16_kernel<<<256, 256, 0, stream>>>(Wq, Wstk, 512 * 512);
  mha_cvt_bf16_kernel<<<256, 256, 0, stream>>>(Wk, Wstk + 512 * 512, 512 * 512);
  mha_cvt_bf16_kernel<<<256, 256, 0, stream>>>(Wv, Wstk + 2 * 512 * 512,
                                               512 * 512);
  mha_cvt_bf16_kernel<<<256, 256, 0, stream>>>(Wp, Wpb, 512 * 512);

  mha_qkv_gemm_kernel<<<dim3(128, 12), 128, 0, stream>>>(Xb, Wstk, Qb, Kb, Vb);
  mha_flash_attn_kernel<<<dim3(SEQ_T / 64, NUM_B * NHEAD), 128, 0, stream>>>(
      Qb, Kb, Vb, Ab);
  mha_out_proj_kernel<<<dim3(128, 4), 128, 0, stream>>>(Ab, Wpb, out);
}
